// LlamaAttention_14147622273780
// MI455X (gfx1250) — matrix operand order repaired and checked by reading, not compiled, not run on hardware
//
#include <hip/hip_runtime.h>
#include <hip/hip_bf16.h>
#include <stdint.h>

// ---------------------------------------------------------------------------
// Llama attention layer on MI455X (gfx1250).
// All matmuls -> v_wmma_f32_16x16x32_bf16 (bf16 in, fp32 accumulate).
// K/V flash-attention tile staging -> Tensor Data Mover (tensor_load_to_lds,
// TENSORcnt) when the toolchain exposes the builtin; cooperative b128 loads
// otherwise.
//
// Roofline: ~480 GFLOP of GEMM-like work vs ~200 MB mandatory HBM traffic
// (23.3 TB/s -> ~10 us), so the layer is WMMA-throughput bound. GEMM uses
// 128x128 C tiles (~64 FLOP per HBM byte) with register double-buffering;
// flash attention shares each 32-key K/V slab across 4 waves (64 Q rows).
// ---------------------------------------------------------------------------

typedef __attribute__((ext_vector_type(16))) __bf16       v16bf;
typedef __attribute__((ext_vector_type(8)))  float        v8f;
typedef __attribute__((ext_vector_type(4)))  unsigned int v4u;
typedef __attribute__((ext_vector_type(8)))  int          v8i;
typedef __attribute__((ext_vector_type(4)))  int          v4i;

#define NH      32
#define KVH     8
#define HD      128
#define S_LEN   2048
#define BATCH   2
#define HID     4096
#define M_TOK   (BATCH * S_LEN)          // 4096 tokens

#if defined(__HIP_DEVICE_COMPILE__) && __has_builtin(__builtin_amdgcn_tensor_load_to_lds)
#define USE_TDM 1
#else
#define USE_TDM 0
#endif

// ---------------------------------------------------------------------------
// fp32 -> bf16 bulk convert
// ---------------------------------------------------------------------------
__global__ void f2bf_kernel(const float* __restrict__ src,
                            __bf16* __restrict__ dst, size_t n)
{
    size_t i = ((size_t)blockIdx.x * blockDim.x + threadIdx.x) * 4;
    if (i + 3 < n) {
        float4 v = *(const float4*)(src + i);
        dst[i + 0] = (__bf16)v.x;
        dst[i + 1] = (__bf16)v.y;
        dst[i + 2] = (__bf16)v.z;
        dst[i + 3] = (__bf16)v.w;
    } else {
        for (; i < n; ++i) dst[i] = (__bf16)src[i];
    }
}

// ---------------------------------------------------------------------------
// TDM: load a 2D bf16 tile (tile_rows x 128 elems) global -> LDS, writing
// rows with stride (128+8) bf16 via the D# pad feature (pad 4 DWORDs after
// every 64 DWORDs). Descriptor packing per CDNA5 ISA 8.3/8.4.
// ---------------------------------------------------------------------------
#if USE_TDM
__device__ __forceinline__ void tdm_load_tile32x128(unsigned lds_addr,
                                                    const void* gptr)
{
    unsigned long long ga = (unsigned long long)(uintptr_t)gptr;
    v4u g0;
    g0[0] = 1u;                                        // count=1, user mode
    g0[1] = lds_addr;                                  // lds_addr
    g0[2] = (unsigned)(ga & 0xFFFFFFFFu);              // global_addr[31:0]
    g0[3] = (unsigned)((ga >> 32) & 0x01FFFFFFu)       // global_addr[56:32]
          | (2u << 30);                                // type = 2 (image)

    const unsigned data_size    = 1u;    // 2-byte elements
    const unsigned pad_enable   = 1u;
    const unsigned pad_interval = 5u;    // 2^(5+1) = 64 DWORDs (one 256B row)
    const unsigned pad_amount   = 3u;    // 3+1 = 4 DWORDs (16B pad)
    v8i g1;
    g1[0] = (int)((data_size << 16) | (pad_enable << 20) |
                  (pad_interval << 22) | (pad_amount << 25));
    g1[1] = (int)(128u << 16);           // tensor_dim0[15:0]=128 @ bits 63:48
    g1[2] = (int)((unsigned)S_LEN << 16);// tensor_dim1[15:0] @ bits 95:80
    g1[3] = (int)(128u << 16);           // tile_dim0=128 @ bits 127:112
    g1[4] = (int)32u;                    // tile_dim1=32, tile_dim2=0
    g1[5] = (int)128u;                   // tensor_dim0_stride = 128 elems
    g1[6] = 0;                           // stride hi / dim1_stride (unused 2D)
    g1[7] = 0;

    v4i z = {0, 0, 0, 0};
#if __clang_major__ >= 23
    v8i z8 = {0, 0, 0, 0, 0, 0, 0, 0};
    __builtin_amdgcn_tensor_load_to_lds(g0, g1, z, z, z8, 0);
#else
    __builtin_amdgcn_tensor_load_to_lds(g0, g1, z, z, 0);
#endif
}
#endif

// ---------------------------------------------------------------------------
// C[M,N] = A[M,K] * B[N,K]^T   (A,B bf16 row-major, C fp32)
// 256 threads = 8 waves; workgroup computes a 128x128 C tile, wave w owns rows
// [16w,16w+16) x 128 cols (8 accumulators). K streamed in 32-wide LDS slabs,
// next slab fetched into registers while WMMAs consume the current one.
// Fragment layouts per CDNA5 ISA 7.12.2 (wave32):
//   A 16x32 bf16 : lane -> M = lane%16 ; K in {0..7,16..23}+8*(lane>=16)
//   B 32x16 bf16 : lane -> N = lane%16 ; K in [0,16)+16*(lane>=16) as loaded here, matching the A load
//   C 16x16 f32  : row = vgpr + 8*(lane>=16), col = lane%16
// ---------------------------------------------------------------------------
__global__ void gemm_bf16_wmma(const __bf16* __restrict__ A,
                               const __bf16* __restrict__ Bw,
                               float* __restrict__ C,
                               int M, int N, int K)
{
    __shared__ __bf16 sA[128][40];   // 32 cols + 8 pad (row = 80B, 16B aligned)
    __shared__ __bf16 sB[128][40];

    const int tid  = threadIdx.x;        // 0..255
    const int wave = tid >> 5;           // 0..7
    const int lane = tid & 31;
    const int row0 = blockIdx.y * 128;
    const int col0 = blockIdx.x * 128;

    const int r    = tid >> 1;           // 0..127
    const int half = (tid & 1) * 16;     // 0 / 16

    const __bf16* gA = A  + (size_t)(row0 + r) * K + half;
    const __bf16* gB = Bw + (size_t)(col0 + r) * K + half;

    // prologue: first slab into registers
    uint4 ra0 = *(const uint4*)(gA);
    uint4 ra1 = *(const uint4*)(gA + 8);
    uint4 rb0 = *(const uint4*)(gB);
    uint4 rb1 = *(const uint4*)(gB + 8);

    v8f acc[8] = {};

    for (int kk = 0; kk < K; kk += 32) {
        *(uint4*)&sA[r][half]     = ra0;
        *(uint4*)&sA[r][half + 8] = ra1;
        *(uint4*)&sB[r][half]     = rb0;
        *(uint4*)&sB[r][half + 8] = rb1;
        __syncthreads();

        if (kk + 32 < K) {               // overlap next slab with compute
            ra0 = *(const uint4*)(gA + kk + 32);
            ra1 = *(const uint4*)(gA + kk + 40);
            rb0 = *(const uint4*)(gB + kk + 32);
            rb1 = *(const uint4*)(gB + kk + 40);
        }

        v16bf afrag;
        {
            int m    = 16 * wave + (lane & 15);
            int koff = (lane < 16) ? 0 : 16;
            #pragma unroll
            for (int j = 0; j < 16; ++j)
                afrag[j] = sA[m][j + koff];
        }
        #pragma unroll
        for (int c = 0; c < 8; ++c) {
            v16bf bfrag;
            int n     = lane & 15;
            int kbase = (lane < 16) ? 0 : 16;
            #pragma unroll
            for (int j = 0; j < 16; ++j)
                bfrag[j] = sB[c * 16 + n][kbase + j];
            acc[c] = __builtin_amdgcn_wmma_f32_16x16x32_bf16(
                false, afrag, false, bfrag, (short)0, acc[c], false, false);
        }
        __syncthreads();
    }

    {
        int n     = lane & 15;
        int rbase = (lane < 16) ? 0 : 8;
        #pragma unroll
        for (int c = 0; c < 8; ++c)
            #pragma unroll
            for (int i = 0; i < 8; ++i) {
                int row = row0 + 16 * wave + rbase + i;
                int col = col0 + c * 16 + n;
                C[(size_t)row * N + col] = acc[c][i];
            }
    }
}

// ---------------------------------------------------------------------------
// RoPE + layout transform: token-major fp32 [tok][nh*HD] ->
// head-major bf16 [b][nh][s][HD]. Angles computed from s (position_ids is
// arange; no need to touch the int64 tensor).
// ---------------------------------------------------------------------------
__global__ void rope_reshape(const float* __restrict__ src,
                             __bf16* __restrict__ dst,
                             int nh, int apply_rope)
{
    size_t idx   = (size_t)blockIdx.x * blockDim.x + threadIdx.x;
    size_t total = (size_t)M_TOK * nh * HD;
    if (idx >= total) return;

    int    d   = (int)(idx & (HD - 1));
    size_t t   = idx >> 7;
    int    hh  = (int)(t % nh);
    size_t tok = t / nh;
    int    s   = (int)(tok % S_LEN);
    int    b   = (int)(tok / S_LEN);

    const float* row = src + tok * ((size_t)nh * HD) + (size_t)hh * HD;
    float x = row[d];
    float val;
    if (apply_rope) {
        int   fi   = d & 63;
        float invf = __expf(-(float)fi * (1.0f / 64.0f) * 9.210340371976184f);
        float ang  = (float)s * invf;
        float c  = __cosf(ang);
        float sn = __sinf(ang);
        if (d < 64) { float x2 = row[d + 64]; val = x * c - x2 * sn; }
        else        { float x1 = row[d - 64]; val = x * c + x1 * sn; }
    } else {
        val = x;
    }
    dst[(((size_t)(b * nh + hh) * S_LEN) + s) * HD + d] = (__bf16)val;
}

// ---------------------------------------------------------------------------
// Causal flash attention. Block = 128 threads = 4 waves = 64 Q rows of one
// (b, h); wave w owns rows [q0b+16w, q0b+16w+16). Each 32-key step:
//   - K/V 32x128 bf16 tiles staged once per block (TDM or cooperative b128)
//   - Q.K^T : 2x4 WMMAs (d chunked by 32)
//   - fp32 online softmax through per-wave LDS slice (doubles as the
//     C-layout -> A-layout transpose for P)
//   - P.V : 8 WMMAs (d chunked by 16) with per-row accumulator rescale.
// Fully-masked steps contribute exactly zero (exp(-1e30-m)=0, factor=1), so
// all waves run all steps and barriers stay uniform.
// ---------------------------------------------------------------------------
__global__ void flash_attn_wmma(const __bf16* __restrict__ Qbf,  // [B][NH][S][HD]
                                const __bf16* __restrict__ Kbf,  // [B][KVH][S][HD]
                                const __bf16* __restrict__ Vbf,  // [B][KVH][S][HD]
                                __bf16* __restrict__ Obf)        // [tok][NH*HD]
{
    __shared__ __bf16 sK[32][HD + 8];
    __shared__ __bf16 sV[32][HD + 8];
    __shared__ float  sScore[4][16][32];
    __shared__ __bf16 sP[4][16][32];
    __shared__ float  rowmax[4][16], rowsum[4][16], rowscale[4][16];

    const int tid  = threadIdx.x;          // 0..127
    const int wave = tid >> 5;             // 0..3
    const int lane = tid & 31;
    const int bid  = blockIdx.x;
    const int qt   = bid & 31;             // S/64 = 32 blocks of Q rows
    const int h    = (bid >> 5) & 31;
    const int b    = bid >> 10;
    const int kh   = h >> 2;               // GQA: 4 query heads per KV head
    const int q0b  = qt * 64;
    const int q0   = q0b + wave * 16;

    const size_t qbase  = (size_t)(b * NH  + h)  * S_LEN * HD;
    const size_t kvbase = (size_t)(b * KVH + kh) * S_LEN * HD;

    // Q fragments: 4 chunks of 32 along d, A-layout
    v16bf aq[4];
    {
        int m    = lane & 15;
        int koff = (lane < 16) ? 0 : 16;
        const __bf16* qrow = Qbf + qbase + (size_t)(q0 + m) * HD;
        #pragma unroll
        for (int c = 0; c < 4; ++c)
            #pragma unroll
            for (int j = 0; j < 16; ++j)
                aq[c][j] = qrow[c * 32 + j + koff];
    }

    if (lane < 16) {
        rowmax[wave][lane]   = -1e30f;
        rowsum[wave][lane]   = 0.0f;
        rowscale[wave][lane] = 0.0f;
    }

    v8f oacc[8] = {};
    const int   nsteps = (q0b + 64) / 32;
    const float scale  = 0.08838834764831845f;      // 1/sqrt(128)

    for (int step = 0; step < nsteps; ++step) {
        const int k0 = step * 32;

        __syncthreads();   // previous step's LDS reads complete before restage
#if USE_TDM
        if (wave == 0) {
            tdm_load_tile32x128((unsigned)(uintptr_t)&sK[0][0],
                                Kbf + kvbase + (size_t)k0 * HD);
            tdm_load_tile32x128((unsigned)(uintptr_t)&sV[0][0],
                                Vbf + kvbase + (size_t)k0 * HD);
            __builtin_amdgcn_s_wait_tensorcnt(0);
        }
#else
        {
            int kr = tid >> 2;             // 0..31 key row
            int c0 = (tid & 3) * 32;       // 0/32/64/96
            const __bf16* krow = Kbf + kvbase + (size_t)(k0 + kr) * HD + c0;
            const __bf16* vrow = Vbf + kvbase + (size_t)(k0 + kr) * HD + c0;
            *(uint4*)&sK[kr][c0]      = *(const uint4*)(krow);
            *(uint4*)&sK[kr][c0 + 8]  = *(const uint4*)(krow + 8);
            *(uint4*)&sK[kr][c0 + 16] = *(const uint4*)(krow + 16);
            *(uint4*)&sK[kr][c0 + 24] = *(const uint4*)(krow + 24);
            *(uint4*)&sV[kr][c0]      = *(const uint4*)(vrow);
            *(uint4*)&sV[kr][c0 + 8]  = *(const uint4*)(vrow + 8);
            *(uint4*)&sV[kr][c0 + 16] = *(const uint4*)(vrow + 16);
            *(uint4*)&sV[kr][c0 + 24] = *(const uint4*)(vrow + 24);
        }
#endif
        __syncthreads();

        // scores: S[16x32] = Q[16x128] . K^T, two 16x16 accumulators
        v8f sa = {}, sb = {};
        #pragma unroll
        for (int c = 0; c < 4; ++c) {
            v16bf bk0, bk1;
            int n     = lane & 15;
            int dbase = (lane < 16) ? 0 : 16;
            #pragma unroll
            for (int j = 0; j < 16; ++j) {
                int dd = c * 32 + dbase + j;
                bk0[j] = sK[n][dd];
                bk1[j] = sK[16 + n][dd];
            }
            sa = __builtin_amdgcn_wmma_f32_16x16x32_bf16(
                false, aq[c], false, bk0, (short)0, sa, false, false);
            sb = __builtin_amdgcn_wmma_f32_16x16x32_bf16(
                false, aq[c], false, bk1, (short)0, sb, false, false);
        }

        // scale + causal mask, spill to this wave's LDS slice
        {
            int n     = lane & 15;
            int rbase = (lane < 16) ? 0 : 8;
            #pragma unroll
            for (int i = 0; i < 8; ++i) {
                int q  = q0 + rbase + i;
                int ka = k0 + n, kb = k0 + 16 + n;
                sScore[wave][rbase + i][n]      = (ka <= q) ? sa[i] * scale : -1e30f;
                sScore[wave][rbase + i][16 + n] = (kb <= q) ? sb[i] * scale : -1e30f;
            }
        }
        __syncthreads();

        // online softmax: lane r (<16) owns row r of its wave
        if (lane < 16) {
            float mold = rowmax[wave][lane];
            float mnew = mold;
            #pragma unroll
            for (int j = 0; j < 32; ++j) mnew = fmaxf(mnew, sScore[wave][lane][j]);
            float fac = __expf(mold - mnew);
            float s   = 0.0f;
            #pragma unroll
            for (int j = 0; j < 32; ++j) {
                float p = __expf(sScore[wave][lane][j] - mnew);
                s += p;
                sP[wave][lane][j] = (__bf16)p;
            }
            rowsum[wave][lane]   = rowsum[wave][lane] * fac + s;
            rowscale[wave][lane] = fac;
            rowmax[wave][lane]   = mnew;
        }
        __syncthreads();

        // rescale accumulators, then O += P.V
        {
            int rbase = (lane < 16) ? 0 : 8;
            float f[8];
            #pragma unroll
            for (int i = 0; i < 8; ++i) f[i] = rowscale[wave][rbase + i];
            #pragma unroll
            for (int c = 0; c < 8; ++c)
                #pragma unroll
                for (int i = 0; i < 8; ++i)
                    oacc[c][i] *= f[i];

            v16bf ap;                     // P 16x32 in A layout (LDS transpose)
            {
                int m    = lane & 15;
                int koff = (lane < 16) ? 0 : 16;
                #pragma unroll
                for (int j = 0; j < 16; ++j)
                    ap[j] = sP[wave][m][j + koff];
            }
            int n     = lane & 15;
            int kbase = (lane < 16) ? 0 : 16;
            #pragma unroll
            for (int c = 0; c < 8; ++c) {
                v16bf bv;                 // V 32x16 chunk in B layout
                #pragma unroll
                for (int j = 0; j < 16; ++j)
                    bv[j] = sV[kbase + j][c * 16 + n];
                oacc[c] = __builtin_amdgcn_wmma_f32_16x16x32_bf16(
                    false, ap, false, bv, (short)0, oacc[c], false, false);
            }
        }
    }

    // finalize: divide by row sums, token-major bf16 for the out-proj GEMM
    {
        int n     = lane & 15;
        int rbase = (lane < 16) ? 0 : 8;
        #pragma unroll
        for (int i = 0; i < 8; ++i) {
            float  inv   = 1.0f / rowsum[wave][rbase + i];
            size_t tok   = (size_t)b * S_LEN + q0 + rbase + i;
            __bf16* orow = Obf + tok * (NH * HD) + h * HD;
            #pragma unroll
            for (int c = 0; c < 8; ++c)
                orow[c * 16 + n] = (__bf16)(oacc[c][i] * inv);
        }
    }
}

// ---------------------------------------------------------------------------
// Host-side orchestration
// ---------------------------------------------------------------------------
extern "C" void kernel_launch(void* const* d_in, const int* in_sizes, int n_in,
                              void* d_out, int out_size, void* d_ws, size_t ws_size,
                              hipStream_t stream)
{
    const float* hs = (const float*)d_in[0];   // [B,S,H] fp32
    const float* wq = (const float*)d_in[1];   // [4096,4096]
    const float* wk = (const float*)d_in[2];   // [1024,4096]
    const float* wv = (const float*)d_in[3];   // [1024,4096]
    const float* wo = (const float*)d_in[4];   // [4096,4096]
    // d_in[5] attention_mask, d_in[6] position_ids: replicated analytically.

    char* ws = (char*)d_ws;
    const size_t o_Xbf  = 0;                                   // 4096x4096 bf16
    const size_t o_Wq   = o_Xbf + (size_t)M_TOK * HID * 2;
    const size_t o_Wk   = o_Wq  + (size_t)HID * HID * 2;
    const size_t o_Wv   = o_Wk  + (size_t)(KVH * HD) * HID * 2;
    const size_t o_Wo   = o_Wv  + (size_t)(KVH * HD) * HID * 2;
    const size_t o_Qf   = o_Wo  + (size_t)HID * HID * 2;       // fp32 4096x4096
    const size_t o_Kf   = o_Qf  + (size_t)M_TOK * HID * 4;
    const size_t o_Vf   = o_Kf  + (size_t)M_TOK * (KVH * HD) * 4;
    const size_t o_Qbf  = o_Vf  + (size_t)M_TOK * (KVH * HD) * 4;
    const size_t o_Kbf  = o_Qbf + (size_t)M_TOK * HID * 2;
    const size_t o_Vbf  = o_Kbf + (size_t)M_TOK * (KVH * HD) * 2;
    const size_t o_Attn = o_Qf;  // alias: Qf dead once rope_reshape(Q) ran

    __bf16* Xbf  = (__bf16*)(ws + o_Xbf);
    __bf16* Wqb  = (__bf16*)(ws + o_Wq);
    __bf16* Wkb  = (__bf16*)(ws + o_Wk);
    __bf16* Wvb  = (__bf16*)(ws + o_Wv);
    __bf16* Wob  = (__bf16*)(ws + o_Wo);
    float*  Qf   = (float*)(ws + o_Qf);
    float*  Kf   = (float*)(ws + o_Kf);
    float*  Vf   = (float*)(ws + o_Vf);
    __bf16* Qbf  = (__bf16*)(ws + o_Qbf);
    __bf16* Kbf  = (__bf16*)(ws + o_Kbf);
    __bf16* Vbf  = (__bf16*)(ws + o_Vbf);
    __bf16* Attn = (__bf16*)(ws + o_Attn);

    auto cvt = [&](const float* s, __bf16* d, size_t n) {
        size_t blocks = (n / 4 + 255) / 256;
        f2bf_kernel<<<dim3((unsigned)blocks), 256, 0, stream>>>(s, d, n);
    };
    cvt(hs, Xbf, (size_t)M_TOK * HID);
    cvt(wq, Wqb, (size_t)HID * HID);
    cvt(wk, Wkb, (size_t)(KVH * HD) * HID);
    cvt(wv, Wvb, (size_t)(KVH * HD) * HID);
    cvt(wo, Wob, (size_t)HID * HID);

    // QKV projections: C[tok, feat] = X . W^T   (128x128 tiles, 256 threads)
    gemm_bf16_wmma<<<dim3(HID / 128, M_TOK / 128), 256, 0, stream>>>(
        Xbf, Wqb, Qf, M_TOK, HID, HID);
    gemm_bf16_wmma<<<dim3((KVH * HD) / 128, M_TOK / 128), 256, 0, stream>>>(
        Xbf, Wkb, Kf, M_TOK, KVH * HD, HID);
    gemm_bf16_wmma<<<dim3((KVH * HD) / 128, M_TOK / 128), 256, 0, stream>>>(
        Xbf, Wvb, Vf, M_TOK, KVH * HD, HID);

    // RoPE + reshape to head-major bf16
    {
        size_t nq = (size_t)M_TOK * HID;
        rope_reshape<<<dim3((unsigned)((nq + 255) / 256)), 256, 0, stream>>>(
            Qf, Qbf, NH, 1);
        size_t nkv = (size_t)M_TOK * KVH * HD;
        rope_reshape<<<dim3((unsigned)((nkv + 255) / 256)), 256, 0, stream>>>(
            Kf, Kbf, KVH, 1);
        rope_reshape<<<dim3((unsigned)((nkv + 255) / 256)), 256, 0, stream>>>(
            Vf, Vbf, KVH, 0);
    }

    // causal flash attention: B * NH * (S/64) blocks of 4 waves
    flash_attn_wmma<<<dim3(BATCH * NH * (S_LEN / 64)), 128, 0, stream>>>(
        Qbf, Kbf, Vbf, Attn);

    // output projection straight into d_out (fp32)
    gemm_bf16_wmma<<<dim3(HID / 128, M_TOK / 128), 256, 0, stream>>>(
        Attn, Wob, (float*)d_out, M_TOK, HID, HID);
}
